// GeniePathLayer_1726576854976
// MI455X (gfx1250) — matrix operand order repaired and checked by reading, not compiled, not run on hardware
//
#include <hip/hip_runtime.h>
#include <hip/hip_bf16.h>
#include <stdint.h>

// ---------------------------------------------------------------------------
// GeniePath layer for MI455X (gfx1250, wave32):
//   GATConv(D=256, heads=1, self-loops) -> tanh -> 1-step LSTM (bias-free)
// GEMMs: v_wmma_f32_16x16x32_bf16 (bf16 in, f32 accumulate).
// Edge softmax/aggregation: by-dst CSR gather (no float atomics).
// gates GEMM: B fragments staged in LDS, shared by 8 row-tile waves.
// ---------------------------------------------------------------------------

#define DIMD 256
#define KTS  8            // 256 / 32 k-steps per 256-deep GEMM
#define NEG_SLOPE 0.2f
#define SCAN_THREADS 1024
#define GL_WAVES 8        // row-tile waves sharing one LDS B-stage

typedef __attribute__((ext_vector_type(16))) __bf16 v16bf;
typedef __attribute__((ext_vector_type(8)))  float  v8f;

__device__ __forceinline__ uint16_t f2bf(float f) {
  uint32_t u = __float_as_uint(f);
  uint32_t r = u + 0x7FFFu + ((u >> 16) & 1u);   // round-to-nearest-even
  return (uint16_t)(r >> 16);
}

__device__ __forceinline__ float lrelu(float a) {
  return a > 0.f ? a : NEG_SLOPE * a;
}

__device__ __forceinline__ float sigmoidf(float x) {
  return 1.0f / (1.0f + __expf(-x));
}

// ---- WMMA fragment helpers -------------------------------------------------
// A (16x32 bf16, row-major source [rows][ldA]):
//   lane = h*16+m holds row m; elems 0..7 = K k0+8h..k0+8h+7,
//   elems 8..15 = K k0+16+8h..k0+16+8h+7  -> two contiguous 16B loads.
__device__ __forceinline__ v16bf load_a_frag(const uint16_t* __restrict__ A,
                                             int rowBase, int k0, int ldA) {
  int lane = threadIdx.x & 31;
  int m = lane & 15, half = lane >> 4;
  const uint16_t* p = A + (size_t)(rowBase + m) * ldA + k0 + half * 8;
  v16bf a;
  uint4* av = reinterpret_cast<uint4*>(&a);
  av[0] = *reinterpret_cast<const uint4*>(p);
  av[1] = *reinterpret_cast<const uint4*>(p + 16);
  return a;
}

// B fragments pre-packed: frag f, lane l -> 16 contiguous bf16 at (f*32+l)*16.
__device__ __forceinline__ v16bf load_b_frag(const uint16_t* __restrict__ Bpk,
                                             int frag) {
  int lane = threadIdx.x & 31;
  return *reinterpret_cast<const v16bf*>(Bpk + ((size_t)frag * 32 + lane) * 16);
}

// Same, but from an LDS stage (frag-contiguous layout, 32B/lane).
__device__ __forceinline__ v16bf load_b_frag_lds(const uint16_t* Bs, int lf) {
  int lane = threadIdx.x & 31;
  return *reinterpret_cast<const v16bf*>(Bs + (size_t)lf * 512 + lane * 16);
}

__device__ __forceinline__ v8f wmma_bf16(v16bf a, v16bf b, v8f c) {
  return __builtin_amdgcn_wmma_f32_16x16x32_bf16(false, a, false, b,
                                                 (short)0, c, false, false);
}

// ---------------------------------------------------------------------------
// K1: pack a weight matrix (logical B[k][n] = W[k*sK + n*sN]) into WMMA
//     B-fragment layout. grid.x = numKT*numNT, block = 32 (one wave).
//     B 32x16 bf16 layout: lane = h*16+n holds column n, elems i<8 -> K = 8h+i, i>=8 -> K = 16+8h+(i-8).
__global__ void pack_weight_frag(const float* __restrict__ W,
                                 uint16_t* __restrict__ dst,
                                 int numNT, int sK, int sN) {
  int frag = blockIdx.x;
  int lane = threadIdx.x;
  int kt = frag / numNT, nt = frag % numNT;
  int n = lane & 15, half = lane >> 4;
  uint16_t* o = dst + ((size_t)frag * 32 + lane) * 16;
  int col = nt * 16 + n;
#pragma unroll
  for (int i = 0; i < 16; ++i) {
    int k = kt * 32 + (i < 8 ? half * 8 + i : 16 + half * 8 + (i - 8));
    o[i] = f2bf(W[(size_t)k * sK + (size_t)col * sN]);
  }
}

// K2: f32 -> bf16 bulk conversion
__global__ void f32_to_bf16_kern(const float* __restrict__ src,
                                 uint16_t* __restrict__ dst, int n) {
  int i = blockIdx.x * blockDim.x + threadIdx.x;
  if (i < n) dst[i] = f2bf(src[i]);
}

// K3: xw = x @ W_gat  (M=N rows, Ncols=256).  grid(rowTiles, 4), block(32,4).
__global__ void gemm_xw(const uint16_t* __restrict__ A,
                        const uint16_t* __restrict__ Bpk,
                        float* __restrict__ C) {
  int rt = blockIdx.x;
  int nt = blockIdx.y * blockDim.y + threadIdx.y;   // 0..15
  int lane = threadIdx.x;
  int rowBase = rt * 16;
  v8f acc = {};
#pragma unroll
  for (int kt = 0; kt < KTS; ++kt) {
    v16bf a = load_a_frag(A, rowBase, kt * 32, DIMD);
    v16bf b = load_b_frag(Bpk, kt * 16 + nt);
    acc = wmma_bf16(a, b, acc);
  }
  // C/D f32 layout: lane = h*16+n, VGPR j -> row 8h+j, col n
  int n = lane & 15, mOff = (lane >> 4) * 8;
  float* out = C + (size_t)(rowBase + mOff) * DIMD + nt * 16 + n;
#pragma unroll
  for (int j = 0; j < 8; ++j) out[(size_t)j * DIMD] = acc[j];
}

// K4: a_s = xw @ att_src, a_d = xw @ att_dst.  One wave per row.
__global__ void attn_logits(const float* __restrict__ xw,
                            const float* __restrict__ a_src,
                            const float* __restrict__ a_dst,
                            float* __restrict__ as_, float* __restrict__ ad_,
                            int N) {
  int row = blockIdx.x * blockDim.y + threadIdx.y;
  if (row >= N) return;
  int lane = threadIdx.x;
  const float* xr = xw + (size_t)row * DIMD;
  float s = 0.f, d = 0.f;
#pragma unroll
  for (int t = 0; t < DIMD / 32; ++t) {
    float v = xr[lane + t * 32];
    s += v * a_src[lane + t * 32];
    d += v * a_dst[lane + t * 32];
  }
#pragma unroll
  for (int off = 16; off; off >>= 1) {
    s += __shfl_down(s, off, 32);
    d += __shfl_down(d, off, 32);
  }
  if (lane == 0) { as_[row] = s; ad_[row] = d; }
}

__device__ __forceinline__ void edge_sd(const int* __restrict__ ei, int E,
                                        int e, int& s, int& d) {
  if (e < E) { s = ei[e]; d = ei[E + e]; }
  else       { s = e - E; d = e - E; }     // self loops appended
}

// K5: zero per-dst degree counters
__global__ void deg_init(int* __restrict__ deg, int N) {
  int i = blockIdx.x * blockDim.x + threadIdx.x;
  if (i < N) deg[i] = 0;
}

// K6: count incoming edges per destination (int atomics only)
__global__ void edge_count(const int* __restrict__ ei, int E, int M,
                           int* __restrict__ deg) {
  int e = blockIdx.x * blockDim.x + threadIdx.x;
  if (e >= M) return;
  int s, d; edge_sd(ei, E, e, s, d);
  (void)s;
  atomicAdd(deg + d, 1);
}

// K7: exclusive scan of degrees -> CSR offsets; zero binning cursors.
//     Single workgroup (1024 threads = 32 waves), N <= SCAN_THREADS * chunk.
__global__ void scan_offsets(const int* __restrict__ deg,
                             int* __restrict__ offs, int* __restrict__ cursor,
                             int N) {
  __shared__ int part[SCAN_THREADS];
  int t = threadIdx.x;
  int chunk = (N + SCAN_THREADS - 1) / SCAN_THREADS;
  int base = t * chunk;
  int sum = 0;
  for (int i = 0; i < chunk; ++i) {
    int idx = base + i;
    if (idx < N) sum += deg[idx];
  }
  part[t] = sum;
  __syncthreads();
  for (int off = 1; off < SCAN_THREADS; off <<= 1) {   // inclusive scan
    int v = (t >= off) ? part[t - off] : 0;
    __syncthreads();
    part[t] += v;
    __syncthreads();
  }
  int run = (t == 0) ? 0 : part[t - 1];                // exclusive base
  for (int i = 0; i < chunk; ++i) {
    int idx = base + i;
    if (idx < N) { offs[idx] = run; cursor[idx] = 0; run += deg[idx]; }
  }
  if (t == SCAN_THREADS - 1) offs[N] = part[SCAN_THREADS - 1];
}

// K8: bin edge sources into CSR slots
__global__ void edge_bin(const int* __restrict__ ei, int E, int M,
                         const int* __restrict__ offs, int* __restrict__ cursor,
                         int* __restrict__ csr_src) {
  int e = blockIdx.x * blockDim.x + threadIdx.x;
  if (e >= M) return;
  int s, d; edge_sd(ei, E, e, s, d);
  int pos = atomicAdd(cursor + d, 1);
  csr_src[offs[d] + pos] = s;
}

// K9: fused GAT aggregation per destination node (one wave per dst):
//     softmax-max, exp/denominator, weighted row gather, +bias, tanh -> bf16.
//     Zero float atomics; source rows stream from L2.
__global__ void gat_gather(const int* __restrict__ offs,
                           const int* __restrict__ csr_src,
                           const float* __restrict__ as_,
                           const float* __restrict__ ad_,
                           const float* __restrict__ xw,
                           const float* __restrict__ bias,
                           uint16_t* __restrict__ xbbf, int N) {
  int d = blockIdx.x * blockDim.y + threadIdx.y;
  if (d >= N) return;
  int lane = threadIdx.x;
  int start = offs[d], cnt = offs[d + 1] - offs[d];    // cnt >= 1 (self loop)
  float add = ad_[d];

  // pass 1: max of leaky-relu logits (lane-parallel + shfl reduce)
  float m = -INFINITY;
  for (int i = lane; i < cnt; i += 32) {
    int s = csr_src[start + i];
    m = fmaxf(m, lrelu(as_[s] + add));
  }
#pragma unroll
  for (int off = 16; off; off >>= 1) m = fmaxf(m, __shfl_xor(m, off, 32));

  // pass 2: exp weights + weighted accumulation of xw rows (8 f32/lane)
  float den = 0.f;
  float acc[DIMD / 32];
#pragma unroll
  for (int t = 0; t < DIMD / 32; ++t) acc[t] = 0.f;
  for (int i = 0; i < cnt; ++i) {
    int s = csr_src[start + i];
    int sn = (i + 1 < cnt) ? csr_src[start + i + 1] : s;
    __builtin_prefetch(xw + (size_t)sn * DIMD + lane * 8, 0, 3);
    float w = __expf(lrelu(as_[s] + add) - m);
    den += w;
    const float* xr = xw + (size_t)s * DIMD;
#pragma unroll
    for (int t = 0; t < DIMD / 32; ++t) acc[t] += w * xr[lane + t * 32];
  }

  // epilogue: normalize, +bias, tanh, emit bf16 row for gates GEMM
  float inv = 1.f / den;
  uint16_t* orow = xbbf + (size_t)d * DIMD;
#pragma unroll
  for (int t = 0; t < DIMD / 32; ++t) {
    int c = lane + t * 32;
    orow[c] = f2bf(tanhf(acc[t] * inv + bias[c]));
  }
}

// K10: fused gates GEMM + LSTM cell, B staged in LDS.
//   gates[n, :] = xb[n] @ W_ih^T + h0[n] @ W_hh^T, split (i|f|g|o) each D wide.
//   grid(ceil(rowTiles/8), 16), block(32,8): one block = one col tile ct,
//   8 waves = 8 row tiles sharing the 64KB of B fragments for that ct.
//   LDS frag order: [mat(2)][kt(8)][gate(4)] -> lf = mat*32 + kt*4 + gate.
__global__ void gates_lstm(const uint16_t* __restrict__ xb,
                           const uint16_t* __restrict__ h0,
                           const uint16_t* __restrict__ Wih_pk,
                           const uint16_t* __restrict__ Whh_pk,
                           const float* __restrict__ c0,
                           float* __restrict__ out, int N, int rowTiles) {
  __shared__ __attribute__((aligned(32))) uint16_t Bs[64 * 512];  // 64 KB
  int ct = blockIdx.y;                         // 0..15, col tile in D
  int lane = threadIdx.x;
  int tid = threadIdx.y * 32 + lane;           // 0..255

  // cooperative stage: 64 frags * 1KB = 4096 uint4, 16 per thread
  for (int i = tid; i < 4096; i += 32 * GL_WAVES) {
    int frag = i >> 6, sub = i & 63;           // 64 uint4 per frag
    int mat = frag >> 5, r = frag & 31;
    int kt = r >> 2, gate = r & 3;
    const uint16_t* g = (mat ? Whh_pk : Wih_pk) +
                        ((size_t)(kt * 64 + gate * 16 + ct) * 512);
    reinterpret_cast<uint4*>(Bs)[i] = reinterpret_cast<const uint4*>(g)[sub];
  }
  __syncthreads();

  int rt = blockIdx.x * GL_WAVES + threadIdx.y;
  if (rt >= rowTiles) return;                  // uniform per-wave guard
  int rowBase = rt * 16;

  v8f ai = {}, af = {}, ag = {}, ao = {};
#pragma unroll
  for (int kt = 0; kt < KTS; ++kt) {           // xb @ W_ih^T
    v16bf a = load_a_frag(xb, rowBase, kt * 32, DIMD);
    ai = wmma_bf16(a, load_b_frag_lds(Bs, kt * 4 + 0), ai);
    af = wmma_bf16(a, load_b_frag_lds(Bs, kt * 4 + 1), af);
    ag = wmma_bf16(a, load_b_frag_lds(Bs, kt * 4 + 2), ag);
    ao = wmma_bf16(a, load_b_frag_lds(Bs, kt * 4 + 3), ao);
  }
#pragma unroll
  for (int kt = 0; kt < KTS; ++kt) {           // + h0 @ W_hh^T
    v16bf a = load_a_frag(h0, rowBase, kt * 32, DIMD);
    ai = wmma_bf16(a, load_b_frag_lds(Bs, 32 + kt * 4 + 0), ai);
    af = wmma_bf16(a, load_b_frag_lds(Bs, 32 + kt * 4 + 1), af);
    ag = wmma_bf16(a, load_b_frag_lds(Bs, 32 + kt * 4 + 2), ag);
    ao = wmma_bf16(a, load_b_frag_lds(Bs, 32 + kt * 4 + 3), ao);
  }

  int n = lane & 15, mOff = (lane >> 4) * 8;
  size_t ND = (size_t)N * DIMD;
#pragma unroll
  for (int j = 0; j < 8; ++j) {
    int row = rowBase + mOff + j;
    size_t idx = (size_t)row * DIMD + ct * 16 + n;
    float iv = sigmoidf(ai[j]);
    float fv = sigmoidf(af[j]);
    float gv = tanhf(ag[j]);
    float ov = sigmoidf(ao[j]);
    float c1 = fv * c0[idx] + iv * gv;
    float h1 = ov * tanhf(c1);
    out[idx]          = h1;   // h1
    out[ND + idx]     = h1;   // h1[None]
    out[2 * ND + idx] = c1;   // c1[None]
  }
}

// ---------------------------------------------------------------------------
extern "C" void kernel_launch(void* const* d_in, const int* in_sizes, int n_in,
                              void* d_out, int out_size, void* d_ws, size_t ws_size,
                              hipStream_t stream) {
  const float* x       = (const float*)d_in[0];
  const int*   ei      = (const int*)d_in[1];    // [2,E] (src row, dst row)
  const float* h0f     = (const float*)d_in[2];
  const float* c0f     = (const float*)d_in[3];
  const float* W_gat   = (const float*)d_in[4];
  const float* att_src = (const float*)d_in[5];
  const float* att_dst = (const float*)d_in[6];
  const float* b_gat   = (const float*)d_in[7];
  const float* W_ih    = (const float*)d_in[8];
  const float* W_hh    = (const float*)d_in[9];
  float* out = (float*)d_out;

  const int N = in_sizes[0] / DIMD;
  const int E = in_sizes[1] / 2;
  const int M = E + N;                 // edges + self loops
  const int ND = N * DIMD;
  const int rowTiles = N / 16;         // N = 10000 -> 625

  // ---- workspace carving (256B aligned) ----
  char* ws = (char*)d_ws;
  size_t off = 0;
  auto carve = [&](size_t bytes) -> char* {
    char* p = ws + off;
    off = (off + bytes + 255) & ~(size_t)255;
    return p;
  };
  uint16_t* wgat_pk = (uint16_t*)carve((size_t)KTS * 16 * 512 * 2);  // 128 KB
  uint16_t* wih_pk  = (uint16_t*)carve((size_t)KTS * 64 * 512 * 2);  // 512 KB
  uint16_t* whh_pk  = (uint16_t*)carve((size_t)KTS * 64 * 512 * 2);  // 512 KB
  uint16_t* xbf     = (uint16_t*)carve((size_t)ND * 2);
  uint16_t* hbf     = (uint16_t*)carve((size_t)ND * 2);
  uint16_t* xbbf    = (uint16_t*)carve((size_t)ND * 2);
  float*    xw      = (float*)carve((size_t)ND * 4);
  float*    as_     = (float*)carve((size_t)N * 4);
  float*    ad_     = (float*)carve((size_t)N * 4);
  int*      deg     = (int*)carve((size_t)N * 4);
  int*      offs    = (int*)carve((size_t)(N + 1) * 4);
  int*      cursor  = (int*)carve((size_t)N * 4);
  int*      csr_src = (int*)carve((size_t)M * 4);
  (void)ws_size; (void)n_in; (void)out_size;

  // 1) pack weights into WMMA B-fragment layout
  pack_weight_frag<<<KTS * 16, 32, 0, stream>>>(W_gat, wgat_pk, 16, DIMD, 1);
  pack_weight_frag<<<KTS * 64, 32, 0, stream>>>(W_ih, wih_pk, 64, 1, DIMD); // B=W_ih^T
  pack_weight_frag<<<KTS * 64, 32, 0, stream>>>(W_hh, whh_pk, 64, 1, DIMD); // B=W_hh^T

  // 2) activations to bf16
  f32_to_bf16_kern<<<(ND + 255) / 256, 256, 0, stream>>>(x,   xbf, ND);
  f32_to_bf16_kern<<<(ND + 255) / 256, 256, 0, stream>>>(h0f, hbf, ND);

  // 3) xw = x @ W_gat (WMMA)
  gemm_xw<<<dim3(rowTiles, 4), dim3(32, 4), 0, stream>>>(xbf, wgat_pk, xw);

  // 4) attention logits per node
  attn_logits<<<(N + 7) / 8, dim3(32, 8), 0, stream>>>(xw, att_src, att_dst, as_, ad_, N);

  // 5) build by-dst CSR (int atomics only), then gather-style edge softmax
  deg_init<<<(N + 255) / 256, 256, 0, stream>>>(deg, N);
  edge_count<<<(M + 255) / 256, 256, 0, stream>>>(ei, E, M, deg);
  scan_offsets<<<1, SCAN_THREADS, 0, stream>>>(deg, offs, cursor, N);
  edge_bin<<<(M + 255) / 256, 256, 0, stream>>>(ei, E, M, offs, cursor, csr_src);
  gat_gather<<<(N + 7) / 8, dim3(32, 8), 0, stream>>>(offs, csr_src, as_, ad_,
                                                      xw, b_gat, xbbf, N);

  // 6) fused gates GEMM (WMMA, LDS-staged B) + LSTM cell -> d_out (h1, h1, c1)
  gates_lstm<<<dim3((rowTiles + GL_WAVES - 1) / GL_WAVES, 16),
               dim3(32, GL_WAVES), 0, stream>>>(
      xbbf, hbf, wih_pk, whh_pk, c0f, out, N, rowTiles);
}
